// sLSTMCell_39006892982944
// MI455X (gfx1250) — matrix operand order repaired and checked by reading, not compiled, not run on hardware
//
#include <hip/hip_runtime.h>
#include <hip/hip_bf16.h>

// sLSTM: T=1024, B=32, I=1024, H=1024, NH=8, HS=128
#define T_  1024
#define B_  32
#define I_  1024
#define NH_ 8
#define HS_ 128
#define N4_ 512            // 4*HS (per-head gate width)
#define N_  4096           // NH * 4 * HS
#define M_  (T_ * B_)      // 32768 rows of the input GEMM

typedef __attribute__((ext_vector_type(16))) __bf16 v16bf;
typedef __attribute__((ext_vector_type(8)))  __bf16 v8bf;
typedef __attribute__((ext_vector_type(8)))  float  v8f;

// ---- fp32 -> bf16 round-to-nearest-even (bit twiddle; no cvt dependence) ----
__device__ __forceinline__ __bf16 f2bf(float f) {
    union { float f; unsigned u; } in; in.f = f;
    unsigned u = in.u;
    u += 0x7FFFu + ((u >> 16) & 1u);
    union { unsigned short s; __bf16 b; } out;
    out.s = (unsigned short)(u >> 16);
    return out.b;
}

__device__ __forceinline__ v8f vzero8() {
    v8f z;
#pragma unroll
    for (int i = 0; i < 8; ++i) z[i] = 0.f;
    return z;
}

// ---- WMMA fragment loaders (contiguous k = 16*half + s, same for A and B), tiles row-major in LDS ----
// A: 16x32 bf16. lane m = lane&15; lanes<16 hold K = 0..15,
// lanes>=16 hold K = 16..31 (one contiguous 16-elem chunk).
__device__ __forceinline__ v16bf ld_a(const __bf16* tile, int stride,
                                      int mBase, int kBase, int lane) {
    const int m = mBase + (lane & 15);
    const int h = (lane >> 4) * 16;
    return *(const v16bf*)(tile + m * stride + kBase + h);
}

// B: 32x16 bf16 where B[k][n] = W[n][k]. Per lane: column n = lane&15,
// K = half*16 + 0..15  ==  16 contiguous elements of row n of W-tile.
__device__ __forceinline__ v16bf ld_b(const __bf16* tile, int stride,
                                      int nBase, int kBase, int lane) {
    const int n = nBase + (lane & 15);
    const int h = (lane >> 4) * 16;
    return *(const v16bf*)(tile + n * stride + kBase + h);
}

__device__ __forceinline__ v8f wmma_bf16(v16bf a, v16bf b, v8f c) {
    return __builtin_amdgcn_wmma_f32_16x16x32_bf16(
        false, a, false, b, (short)0, c, false, false);
}

// =====================================================================
// Kernel 1: G[t*B+b][n] = bias[n] + sum_i x[t][b][i] * Wih[n][i]
// 128x128 block tile, 256 threads (8 waves), wave tile 64x32,
// K staged 32-wide, double-buffered bf16 LDS.
// =====================================================================
__global__ __launch_bounds__(256) void slstm_gemm_ih(
    const float* __restrict__ X,    // [M_][I_]
    const float* __restrict__ W,    // [N_][I_]
    const float* __restrict__ bias, // [N_]
    float* __restrict__ G)          // [M_][N_]
{
    __shared__ alignas(32) __bf16 Xs[2][128 * 32];
    __shared__ alignas(32) __bf16 Ws[2][128 * 32];

    const int bm = blockIdx.x * 128;
    const int bn = blockIdx.y * 128;
    const int tid = threadIdx.x;
    const int lane = tid & 31;
    const int wave = tid >> 5;
    const int mW = (wave >> 2) * 64;  // 0 / 64
    const int nW = (wave & 3) * 32;   // 0,32,64,96

    v8f acc[4][2];
#pragma unroll
    for (int i = 0; i < 4; ++i)
#pragma unroll
        for (int j = 0; j < 2; ++j) acc[i][j] = vzero8();

    const int sr = tid >> 3;        // 0..31
    const int sc = (tid & 7) * 4;   // 0..28

    auto stage = [&](int buf, int k0) {
#pragma unroll
        for (int p = 0; p < 4; ++p) {
            const int row = sr + p * 32;
            const float4 xv = *(const float4*)(X + (long)(bm + row) * I_ + k0 + sc);
            const float4 wv = *(const float4*)(W + (long)(bn + row) * I_ + k0 + sc);
            __bf16* xd = &Xs[buf][row * 32 + sc];
            xd[0] = f2bf(xv.x); xd[1] = f2bf(xv.y);
            xd[2] = f2bf(xv.z); xd[3] = f2bf(xv.w);
            __bf16* wd = &Ws[buf][row * 32 + sc];
            wd[0] = f2bf(wv.x); wd[1] = f2bf(wv.y);
            wd[2] = f2bf(wv.z); wd[3] = f2bf(wv.w);
        }
    };

    stage(0, 0);
    __syncthreads();

    for (int k0 = 0; k0 < I_; k0 += 32) {
        const int buf = (k0 >> 5) & 1;
        if (k0 + 32 < I_) {
            stage(buf ^ 1, k0 + 32);
            if (k0 + 64 < I_) {   // gfx1250 global_prefetch_b8 on the next-next chunk
                __builtin_prefetch(X + (long)(bm + sr) * I_ + k0 + 64 + sc, 0, 0);
                __builtin_prefetch(W + (long)(bn + sr) * I_ + k0 + 64 + sc, 0, 0);
            }
        }
        v16bf a[4], b[2];
#pragma unroll
        for (int i = 0; i < 4; ++i) a[i] = ld_a(Xs[buf], 32, mW + i * 16, 0, lane);
#pragma unroll
        for (int j = 0; j < 2; ++j) b[j] = ld_b(Ws[buf], 32, nW + j * 16, 0, lane);
#pragma unroll
        for (int i = 0; i < 4; ++i)
#pragma unroll
            for (int j = 0; j < 2; ++j)
                acc[i][j] = wmma_bf16(a[i], b[j], acc[i][j]);
        __syncthreads();
    }

    const int half8 = (lane >> 4) * 8;
#pragma unroll
    for (int j = 0; j < 2; ++j) {
        const int ng = bn + nW + j * 16 + (lane & 15);
        const float bv = bias[ng];
#pragma unroll
        for (int i = 0; i < 4; ++i) {
#pragma unroll
            for (int r = 0; r < 8; ++r) {
                const int mg = bm + mW + i * 16 + r + half8;
                G[(long)mg * N_ + ng] = acc[i][j][r] + bv;
            }
        }
    }
}

// =====================================================================
// Kernel 2: persistent recurrence. 8 blocks (1 per head), 512 threads
// (16 waves). Whh head (bf16, 128 KB) + h-state (bf16, 8 KB) + lin
// staging (fp32, 64 KB) live in the WGP's 320 KB LDS. c/m/n states in
// registers. B (Whh) WMMA fragments are hoisted into persistent VGPRs
// outside the time loop; only the h-state A-fragments reload per step.
// =====================================================================
__global__ __launch_bounds__(512) void slstm_recur(
    const float* __restrict__ Whh,  // [NH_][N4_][HS_]
    const float* __restrict__ G,    // [T_][B_][NH_][N4_]
    float* __restrict__ Out)        // [T_][B_][NH_*HS_]
{
    __shared__ alignas(32) __bf16 Wt[N4_ * HS_];   // 128 KB  (row n, col k)
    __shared__ alignas(32) __bf16 Ah[B_ * HS_];    //   8 KB  (row b, col k)
    __shared__ alignas(32) float  L[B_ * N4_];     //  64 KB  lin_hh staging

    const int hd = blockIdx.x;
    const int tid = threadIdx.x;
    const int lane = tid & 31;
    const int wave = tid >> 5;

    // Stage + convert this head's Whh into LDS (bf16).
    const float* wsrc = Whh + (long)hd * N4_ * HS_;
    for (int e = tid; e < N4_ * HS_; e += 512) Wt[e] = f2bf(wsrc[e]);
    for (int e = tid; e < B_ * HS_; e += 512) Ah[e] = f2bf(0.f);

    // Per-thread state slice: 8 consecutive d for one batch row.
    const int b_el = tid >> 4;          // 0..31
    const int d0   = (tid & 15) * 8;    // 0..120
    float cs[8], ns[8], ms[8];
#pragma unroll
    for (int j = 0; j < 8; ++j) { cs[j] = 0.f; ns[j] = 0.f; ms[j] = 0.f; }

    // GEMM tiling: C is 32x512 => 2 m-tiles x 32 n-tiles; wave w owns
    // n-tiles {2w, 2w+1} and both m-tiles.
    const int nt0 = wave * 2;
    const int half8 = (lane >> 4) * 8;

    __syncthreads();

    // Hoist the 8 Whh fragments (2 n-tiles x 4 k-chunks) into registers:
    // constant across all 1024 timesteps.
    v16bf bf[2][4];
#pragma unroll
    for (int kt = 0; kt < 4; ++kt) {
        bf[0][kt] = ld_b(Wt, HS_, nt0 * 16,       kt * 32, lane);
        bf[1][kt] = ld_b(Wt, HS_, (nt0 + 1) * 16, kt * 32, lane);
    }

    for (int t = 0; t < T_; ++t) {
        v8f acc00 = vzero8(), acc01 = vzero8();
        v8f acc10 = vzero8(), acc11 = vzero8();
#pragma unroll
        for (int kt = 0; kt < 4; ++kt) {
            const int kB = kt * 32;
            v16bf a0 = ld_a(Ah, HS_, 0,  kB, lane);
            v16bf a1 = ld_a(Ah, HS_, 16, kB, lane);
            acc00 = wmma_bf16(a0, bf[0][kt], acc00);
            acc10 = wmma_bf16(a1, bf[0][kt], acc10);
            acc01 = wmma_bf16(a0, bf[1][kt], acc01);
            acc11 = wmma_bf16(a1, bf[1][kt], acc11);
        }
        // Scatter lin_hh into LDS (m = r + half8, n = lane&15 per C layout).
        {
            const int n0 = nt0 * 16 + (lane & 15);
            const int n1 = n0 + 16;
#pragma unroll
            for (int r = 0; r < 8; ++r) {
                const int m0 = r + half8;
                const int m1 = 16 + r + half8;
                L[m0 * N4_ + n0] = acc00[r];
                L[m0 * N4_ + n1] = acc01[r];
                L[m1 * N4_ + n0] = acc10[r];
                L[m1 * N4_ + n1] = acc11[r];
            }
        }
        __syncthreads();

        // Fused sLSTM gate math; add precomputed x-path gates from global.
        const float* g = G + (long)t * B_ * N_ + (long)b_el * N_ + hd * N4_;
        const float* lrow = &L[b_el * N4_];
        float* orow = Out + (long)t * B_ * (NH_ * HS_) + (long)b_el * (NH_ * HS_) + hd * HS_;

        // Prefetch next step's gate slice (gfx1250 global_prefetch_b8).
        if (t + 1 < T_) {
            const float* gn = g + (long)B_ * N_;
            __builtin_prefetch(gn + d0, 0, 0);
            __builtin_prefetch(gn + HS_ + d0, 0, 0);
            __builtin_prefetch(gn + 2 * HS_ + d0, 0, 0);
            __builtin_prefetch(gn + 3 * HS_ + d0, 0, 0);
        }

#pragma unroll
        for (int j = 0; j < 8; ++j) {
            const int d = d0 + j;
            const float iv = lrow[d]             + g[d];
            const float fv = lrow[HS_ + d]       + g[HS_ + d];
            const float zv = lrow[2 * HS_ + d]   + g[2 * HS_ + d];
            const float ov = lrow[3 * HS_ + d]   + g[3 * HS_ + d];
            const float z  = tanhf(zv);
            const float o  = 1.f / (1.f + expf(-ov));   // sigmoid
            const float mn = fmaxf(fv + ms[j], iv);
            const float ie = expf(iv - mn);
            const float fe = expf(fv + ms[j] - mn);
            const float cn = fe * cs[j] + ie * z;
            const float nn = fe * ns[j] + ie;
            const float h  = o * (cn / nn);
            cs[j] = cn; ns[j] = nn; ms[j] = mn;
            orow[d] = h;
            Ah[b_el * HS_ + d] = f2bf(h);
        }
        __syncthreads();  // Ah visible & L free before next step's GEMM
    }
}

extern "C" void kernel_launch(void* const* d_in, const int* in_sizes, int n_in,
                              void* d_out, int out_size, void* d_ws, size_t ws_size,
                              hipStream_t stream) {
    const float* xs   = (const float*)d_in[0];  // [T,B,I]
    const float* wih  = (const float*)d_in[1];  // [NH,4HS,I]
    const float* whh  = (const float*)d_in[2];  // [NH,4HS,HS]
    const float* bias = (const float*)d_in[3];  // [NH,4HS]
    float* out = (float*)d_out;                 // [T,B,H]
    float* G   = (float*)d_ws;                  // [M_][N_] fp32 (512 MB)

    dim3 g1(M_ / 128, N_ / 128);  // 256 x 32
    slstm_gemm_ih<<<g1, 256, 0, stream>>>(xs, wih, bias, G);
    slstm_recur<<<NH_, 512, 0, stream>>>(whh, G, out);
    (void)in_sizes; (void)n_in; (void)out_size; (void)ws_size;
}
